// LSTM_52553219834397
// MI455X (gfx1250) — matrix operand order repaired and checked by reading, not compiled, not run on hardware
//
#include <hip/hip_runtime.h>

typedef __attribute__((ext_vector_type(16))) _Float16 v16h;
typedef __attribute__((ext_vector_type(8)))  float    v8f;

#define HID 50     // hidden size layer 1
#define NG  204    // 200 layer-1 gate rows + 4 layer-2 input-projection rows
#define XCH 256    // x staging chunk (LDS double buffer)

__device__ __forceinline__ float fast_sigmoid(float x) {
    x = fminf(fmaxf(x, -30.f), 30.f);
    return __builtin_amdgcn_rcpf(1.f + __expf(-x));
}
__device__ __forceinline__ float fast_tanh(float x) {
    x = fminf(fmaxf(x, -15.f), 15.f);
    float e = __expf(-2.f * x);                       // v_exp_f32
    return (1.f - e) * __builtin_amdgcn_rcpf(1.f + e);
}

__global__ __launch_bounds__(256, 1)
void lstm2_persistent(const float* __restrict__ x,
                      const float* __restrict__ Wih1, const float* __restrict__ Whh1,
                      const float* __restrict__ bih1, const float* __restrict__ bhh1,
                      const float* __restrict__ Wih2, const float* __restrict__ Whh2,
                      const float* __restrict__ bih2, const float* __restrict__ bhh2,
                      float* __restrict__ out, int T)
{
    __shared__ float gates[256];     // 16 tiles x 16 cols (rows 204..255 = pad)
    __shared__ float hbuf[64];       // h1 (50 real + zero pad to 64)
    __shared__ float xbuf[2][XCH];   // double-buffered input staging

    const int tid  = threadIdx.x;
    const int lane = tid & 31;
    const int wv   = tid >> 5;       // wave id 0..7

    // ---- loop-invariant B fragments (f16) in VGPRs: B[k][n] = Wbig[n][k] ----
    // 32x16 f16 B layout: lane n (0..15) holds K = kc*32 + 0..15 in halfs 0..15,
    // lane 16+n holds K = kc*32 + 16..31. Wave wv owns tiles wv and wv+8.
    auto bfrag = [&](int tile, int kc) -> v16h {
        v16h b = {};
        const int n  = tile * 16 + (lane & 15);
        const int kb = kc * 32 + ((lane & 16) ? 16 : 0);
#pragma unroll
        for (int i = 0; i < 16; ++i) {
            const int k = kb + i;
            float w = 0.f;
            if (k < HID && n < NG)
                w = (n < 200) ? Whh1[n * HID + k] : Wih2[(n - 200) * HID + k];
            b[i] = (_Float16)w;
        }
        return b;
    };
    const v16h bA0 = bfrag(wv, 0),     bA1 = bfrag(wv, 1);
    const v16h bB0 = bfrag(wv + 8, 0), bB1 = bfrag(wv + 8, 1);

    // ---- per-thread resident state ----
    float wi_i = 0.f, wi_f = 0.f, wi_g = 0.f, wi_o = 0.f;
    float bi_i = 0.f, bi_f = 0.f, bi_g = 0.f, bi_o = 0.f;
    float c1 = 0.f;
    if (tid < HID) {
        wi_i = Wih1[tid];           wi_f = Wih1[HID + tid];
        wi_g = Wih1[2 * HID + tid]; wi_o = Wih1[3 * HID + tid];
        bi_i = bih1[tid]           + bhh1[tid];
        bi_f = bih1[HID + tid]     + bhh1[HID + tid];
        bi_g = bih1[2 * HID + tid] + bhh1[2 * HID + tid];
        bi_o = bih1[3 * HID + tid] + bhh1[3 * HID + tid];
    }
    float w2[4] = {0.f, 0.f, 0.f, 0.f}, b2[4] = {0.f, 0.f, 0.f, 0.f};
    float c2 = 0.f, h2 = 0.f;
    if (tid == HID) {
#pragma unroll
        for (int g = 0; g < 4; ++g) { w2[g] = Whh2[g]; b2[g] = bih2[g] + bhh2[g]; }
    }

    if (tid < 64) hbuf[tid] = 0.f;                       // h1_{-1} = 0, pad = 0
    xbuf[0][tid] = (tid < T) ? x[tid] : 0.f;             // stage chunk 0
    __syncthreads();

    // ---- main sequential loop (iteration T is the layer-2 pipeline tail) ----
    for (int t = 0; t <= T; ++t) {
        // A fragment: row M=0 = h1_prev (f16); all other rows zero.
        v16h a0 = {}, a1 = {};
        if (lane == 0) {
#pragma unroll
            for (int i = 0; i < 8; ++i) {
                a0[i]     = (_Float16)hbuf[i];      a0[8 + i] = (_Float16)hbuf[8 + i];
                a1[i]     = (_Float16)hbuf[32 + i]; a1[8 + i] = (_Float16)hbuf[40 + i];
            }
        } else if (lane == 16) {
#pragma unroll
            for (int i = 0; i < 8; ++i) {
                a0[i]     = (_Float16)hbuf[16 + i]; a0[8 + i] = (_Float16)hbuf[24 + i];
                a1[i]     = (_Float16)hbuf[48 + i]; a1[8 + i] = (_Float16)hbuf[56 + i];
            }
        }
        // 204x50 matvec: 2 K-chunks x 2 tiles per wave, f32 accumulate.
        v8f acc = {};
        acc = __builtin_amdgcn_wmma_f32_16x16x32_f16(false, a0, false, bA0, (short)0, acc, false, false);
        acc = __builtin_amdgcn_wmma_f32_16x16x32_f16(false, a1, false, bA1, (short)0, acc, false, false);
        v8f acc2 = {};
        acc2 = __builtin_amdgcn_wmma_f32_16x16x32_f16(false, a0, false, bB0, (short)0, acc2, false, false);
        acc2 = __builtin_amdgcn_wmma_f32_16x16x32_f16(false, a1, false, bB1, (short)0, acc2, false, false);
        if (lane < 16) {                                 // D row M=0 lives in elem 0, lanes 0..15
            gates[wv * 16 + lane]       = acc[0];
            gates[(wv + 8) * 16 + lane] = acc2[0];
        }
        __syncthreads();

        // Layer-1 cell update for time t (thread j owns c1[j]).
        if (t < T && tid < HID) {
            const float xt = xbuf[(t >> 8) & 1][t & (XCH - 1)];
            const float pi = gates[tid]           + fmaf(xt, wi_i, bi_i);
            const float pf = gates[HID + tid]     + fmaf(xt, wi_f, bi_f);
            const float pg = gates[2 * HID + tid] + fmaf(xt, wi_g, bi_g);
            const float po = gates[3 * HID + tid] + fmaf(xt, wi_o, bi_o);
            const float ig = fast_sigmoid(pi), fg = fast_sigmoid(pf);
            const float gg = fast_tanh(pg),    og = fast_sigmoid(po);
            c1 = fmaf(fg, c1, ig * gg);
            hbuf[tid] = og * fast_tanh(c1);
        }
        // Layer-2 cell update for time t-1 (rows 200..203 = W_ih2 @ h1_{t-1}).
        if (tid == HID && t > 0) {
            const float ig = fast_sigmoid(fmaf(h2, w2[0], gates[200] + b2[0]));
            const float fg = fast_sigmoid(fmaf(h2, w2[1], gates[201] + b2[1]));
            const float gg = fast_tanh   (fmaf(h2, w2[2], gates[202] + b2[2]));
            const float og = fast_sigmoid(fmaf(h2, w2[3], gates[203] + b2[3]));
            c2 = fmaf(fg, c2, ig * gg);
            h2 = og * fast_tanh(c2);
            if (t == T) out[0] = h2;                     // h2[T-1]
        }
        // Stage next x chunk into the idle buffer (waves 6-7, off critical path).
        if (((t & (XCH - 1)) == 0) && (t + XCH) < T && tid >= 192) {
            const int base = t + XCH;
            float* dst = xbuf[((t >> 8) + 1) & 1];
#pragma unroll
            for (int r = 0; r < 4; ++r) {
                const int idx = (tid - 192) * 4 + r;
                dst[idx] = (base + idx < T) ? x[base + idx] : 0.f;
            }
        }
        __syncthreads();
    }
}

extern "C" void kernel_launch(void* const* d_in, const int* in_sizes, int n_in,
                              void* d_out, int out_size, void* d_ws, size_t ws_size,
                              hipStream_t stream) {
    (void)n_in; (void)out_size; (void)d_ws; (void)ws_size;
    const float* x    = (const float*)d_in[0];
    const float* Wih1 = (const float*)d_in[1];
    const float* Whh1 = (const float*)d_in[2];
    const float* bih1 = (const float*)d_in[3];
    const float* bhh1 = (const float*)d_in[4];
    const float* Wih2 = (const float*)d_in[5];
    const float* Whh2 = (const float*)d_in[6];
    const float* bih2 = (const float*)d_in[7];
    const float* bhh2 = (const float*)d_in[8];
    const int T = in_sizes[0];   // input_seq is [T, 1]

    lstm2_persistent<<<1, 256, 0, stream>>>(x, Wih1, Whh1, bih1, bhh1,
                                            Wih2, Whh2, bih2, bhh2,
                                            (float*)d_out, T);
}
